// Dy_GrEncoder_80719615361185
// MI455X (gfx1250) — compile-verified, matrix operand order checked by reading, not run on hardware
//
#include <hip/hip_runtime.h>
#include <hip/hip_bf16.h>

// ---------------------------------------------------------------------------
// MI455X (gfx1250) GatedGraphConv + GRUCell + LSTM + Linear fused pipeline.
// wave32; one wave owns a 16-node tile; all dense matmuls via
// v_wmma_f32_16x16x32_f16 (f16 inputs, f32 accumulate); activations via
// hardware V_TANH_F32.
// ---------------------------------------------------------------------------

typedef __attribute__((ext_vector_type(16))) _Float16 v16h;
typedef __attribute__((ext_vector_type(8)))  float    v8f;

#define HID 64

__device__ __forceinline__ v8f wmma_f16(v16h a, v16h b, v8f c) {
  // D = A(16x32 f16) * B(32x16 f16) + C(16x16 f32)
  return __builtin_amdgcn_wmma_f32_16x16x32_f16(
      /*neg_a=*/false, a, /*neg_b=*/false, b,
      /*c_mod=*/(short)0, c, /*reuse_a=*/false, /*reuse_b=*/false);
}

// Hardware tanh (CDNA5 V_TANH_F32, TRANS unit — co-executes with WMMA).
__device__ __forceinline__ float htanh(float x) {
#if __has_builtin(__builtin_amdgcn_tanhf)
  return __builtin_amdgcn_tanhf(x);
#else
  float y;
  // TRANS-op hazard rule: pad before result use / source overwrite.
  asm volatile("v_tanh_f32 %0, %1\n\tv_nop\n\tv_nop" : "=v"(y) : "v"(x));
  return y;
#endif
}

__device__ __forceinline__ float sigm(float v) {
  return 0.5f + 0.5f * htanh(0.5f * v);
}

// K-mapping shared by 16-bit A and B fragments (ISA 7.12.2):
//   element j in 0..7  -> K = kbase + j        (kbase = kstep*32 + (lane>=16?8:0))
//   element j in 8..15 -> K = kbase + 16 + (j-8)

union H8 { uint4 u; _Float16 h[8]; };

// B fragment from f16 weights stored as Wt[n][k], row-major 64 halves per row.
__device__ __forceinline__ v16h b_frag(const _Float16* __restrict__ Wt,
                                       int rowcol, int kbase) {
  const _Float16* p = Wt + rowcol * HID + kbase;
  H8 lo, hi;
  lo.u = *(const uint4*)(p);
  hi.u = *(const uint4*)(p + 16);
  v16h f;
#pragma unroll
  for (int j = 0; j < 8; ++j) { f[j] = lo.h[j]; f[j + 8] = hi.h[j]; }
  return f;
}

// A fragment from an LDS tile (16 rows x 64 f32, row-major).
__device__ __forceinline__ v16h a_frag_lds(const float* __restrict__ base,
                                           int row, int kbase) {
  const float* p = base + row * HID + kbase;
  float4 q0 = *(const float4*)(p);
  float4 q1 = *(const float4*)(p + 4);
  float4 q2 = *(const float4*)(p + 16);
  float4 q3 = *(const float4*)(p + 20);
  v16h f;
  f[0]  = (_Float16)q0.x; f[1]  = (_Float16)q0.y; f[2]  = (_Float16)q0.z; f[3]  = (_Float16)q0.w;
  f[4]  = (_Float16)q1.x; f[5]  = (_Float16)q1.y; f[6]  = (_Float16)q1.z; f[7]  = (_Float16)q1.w;
  f[8]  = (_Float16)q2.x; f[9]  = (_Float16)q2.y; f[10] = (_Float16)q2.z; f[11] = (_Float16)q2.w;
  f[12] = (_Float16)q3.x; f[13] = (_Float16)q3.y; f[14] = (_Float16)q3.z; f[15] = (_Float16)q3.w;
  return f;
}

// A fragment straight from global memory (one node row), scaled (for agg/cnt).
__device__ __forceinline__ v16h a_frag_global_scaled(const float* __restrict__ p,
                                                     int kbase, float s) {
  const float* q = p + kbase;
  float4 q0 = *(const float4*)(q);
  float4 q1 = *(const float4*)(q + 4);
  float4 q2 = *(const float4*)(q + 16);
  float4 q3 = *(const float4*)(q + 20);
  v16h f;
  f[0]  = (_Float16)(q0.x * s); f[1]  = (_Float16)(q0.y * s);
  f[2]  = (_Float16)(q0.z * s); f[3]  = (_Float16)(q0.w * s);
  f[4]  = (_Float16)(q1.x * s); f[5]  = (_Float16)(q1.y * s);
  f[6]  = (_Float16)(q1.z * s); f[7]  = (_Float16)(q1.w * s);
  f[8]  = (_Float16)(q2.x * s); f[9]  = (_Float16)(q2.y * s);
  f[10] = (_Float16)(q2.z * s); f[11] = (_Float16)(q2.w * s);
  f[12] = (_Float16)(q3.x * s); f[13] = (_Float16)(q3.y * s);
  f[14] = (_Float16)(q3.z * s); f[15] = (_Float16)(q3.w * s);
  return f;
}

// ---------------------------------------------------------------------------
// Kernel 0: convert weights to f16 in B-fragment layout Wt[n][k]; bsum = lstm biases.
// ---------------------------------------------------------------------------
__global__ void prep_kernel(const float* __restrict__ Wc,
                            const float* __restrict__ gih,
                            const float* __restrict__ ghh,
                            const float* __restrict__ lih,
                            const float* __restrict__ lin,
                            const float* __restrict__ lbi,
                            const float* __restrict__ lbh,
                            _Float16* __restrict__ wtc,
                            _Float16* __restrict__ wtgih,
                            _Float16* __restrict__ wtghh,
                            _Float16* __restrict__ wtlih,
                            _Float16* __restrict__ wtlin,
                            float* __restrict__ bsum) {
  int t = blockIdx.x * blockDim.x + threadIdx.x;
  int stride = gridDim.x * blockDim.x;
  // conv used as x @ W_conv  ->  Wt[n][k] = W_conv[k][n]  (transpose)
  for (int i = t; i < HID * HID; i += stride) {
    int n = i / HID, k = i % HID;
    wtc[i] = (_Float16)Wc[k * HID + n];
  }
  // the rest are used as x @ W.T -> Wt[n][k] = W[n][k] (straight convert)
  for (int i = t; i < 3 * HID * HID; i += stride) wtgih[i] = (_Float16)gih[i];
  for (int i = t; i < 3 * HID * HID; i += stride) wtghh[i] = (_Float16)ghh[i];
  for (int i = t; i < 4 * HID * HID; i += stride) wtlih[i] = (_Float16)lih[i];
  for (int i = t; i < HID * HID; i += stride)     wtlin[i] = (_Float16)lin[i];
  for (int i = t; i < 4 * HID; i += stride)       bsum[i] = lbi[i] + lbh[i];
}

// ---------------------------------------------------------------------------
// Kernel 1: zero agg + cnt.
// ---------------------------------------------------------------------------
__global__ void zero_kernel(float* __restrict__ agg, float* __restrict__ cnt,
                            int n_nodes) {
  long long total = (long long)n_nodes * HID + n_nodes;
  for (long long i = blockIdx.x * (long long)blockDim.x + threadIdx.x;
       i < total; i += (long long)gridDim.x * blockDim.x) {
    if (i < (long long)n_nodes * HID) agg[i] = 0.0f;
    else cnt[i - (long long)n_nodes * HID] = 0.0f;
  }
}

// ---------------------------------------------------------------------------
// Kernel 2: m = x @ W_conv via WMMA. One wave = 16 nodes.
// ---------------------------------------------------------------------------
__global__ __launch_bounds__(256) void conv_kernel(
    const float* __restrict__ x, const _Float16* __restrict__ wtc,
    float* __restrict__ m, int ntiles) {
  const int lane = threadIdx.x & 31;
  const int w    = threadIdx.x >> 5;
  const int tile = blockIdx.x * 8 + w;
  if (tile >= ntiles) return;
  const int base = tile * 16;

  __shared__ float xs_all[8][16 * HID];
  float* xs = xs_all[w];

  // cooperative load of 16 rows x 64 f32 by this wave
  const float4* src = (const float4*)(x + (size_t)base * HID);
#pragma unroll
  for (int i = lane; i < 16 * HID / 4; i += 32) ((float4*)xs)[i] = src[i];

  const int row  = lane & 15;
  const int hi8  = (lane & 16) ? 8 : 0;
  const int mrow0 = (lane >> 4) * 8;

  v16h a0 = a_frag_lds(xs, row, hi8);
  v16h a1 = a_frag_lds(xs, row, 32 + hi8);

#pragma unroll
  for (int t = 0; t < 4; ++t) {
    const int col = t * 16 + row;
    v16h b0 = b_frag(wtc, col, hi8);
    v16h b1 = b_frag(wtc, col, 32 + hi8);
    v8f acc = {};
    acc = wmma_f16(a0, b0, acc);
    acc = wmma_f16(a1, b1, acc);
#pragma unroll
    for (int v = 0; v < 8; ++v)
      m[(size_t)(base + mrow0 + v) * HID + col] = acc[v];
  }
}

// ---------------------------------------------------------------------------
// Kernel 3: edge scatter: agg[dst] += m[src]; cnt[dst] += 1.
// One thread per (edge, column): each wave issues contiguous 128B atomic
// bursts into one destination row; m[src] gather reads are contiguous too.
// ---------------------------------------------------------------------------
__global__ __launch_bounds__(256) void scatter_kernel(
    const int* __restrict__ src, const int* __restrict__ dst,
    const float* __restrict__ m, float* __restrict__ agg,
    float* __restrict__ cnt, int n_edges) {
  long long tid = blockIdx.x * (long long)blockDim.x + threadIdx.x;
  if (tid >= (long long)n_edges * HID) return;
  int e = (int)(tid >> 6);
  int h = (int)(tid & (HID - 1));
  int s = src[e], d = dst[e];
  atomicAdd(agg + (size_t)d * HID + h, m[(size_t)s * HID + h]);
  if (h == 0) atomicAdd(cnt + d, 1.0f);
}

// ---------------------------------------------------------------------------
// Kernel 4: fused per-node pipeline: normalize -> GRU -> LSTM -> ReLU -> Linear.
// One wave = 16 nodes; 3 private LDS buffers (x->tanhC, r->hconv->relu, z->sigI).
// ---------------------------------------------------------------------------
__global__ __launch_bounds__(128) void node_kernel(
    const float* __restrict__ x, const float* __restrict__ agg,
    const float* __restrict__ cnt,
    const _Float16* __restrict__ wtgih, const _Float16* __restrict__ wtghh,
    const _Float16* __restrict__ wtlih, const _Float16* __restrict__ wtlin,
    const float* __restrict__ gbi, const float* __restrict__ gbh,
    const float* __restrict__ bsum, const float* __restrict__ linb,
    float* __restrict__ out, int ntiles) {
  const int lane = threadIdx.x & 31;
  const int w    = threadIdx.x >> 5;
  const int tile = blockIdx.x * 4 + w;
  if (tile >= ntiles) return;
  const int base = tile * 16;

  __shared__ float smem[4 * 3 * 16 * HID];   // 48 KB
  float* xs = smem + w * 3 * 16 * HID;       // x tile, later tanh(c)
  float* bA = xs + 16 * HID;                 // r, later h_conv, later relu(h~)
  float* bB = bA + 16 * HID;                 // z, later sig(i)

  // load x tile into LDS (needed again for h_conv combine)
  const float4* srcv = (const float4*)(x + (size_t)base * HID);
#pragma unroll
  for (int i = lane; i < 16 * HID / 4; i += 32) ((float4*)xs)[i] = srcv[i];

  const int row   = lane & 15;
  const int hi8   = (lane & 16) ? 8 : 0;
  const int mrow0 = (lane >> 4) * 8;

  // A fragments: x (for gh) and normalized agg (for gi)
  v16h xa0 = a_frag_lds(xs, row, hi8);
  v16h xa1 = a_frag_lds(xs, row, 32 + hi8);
  float cv = cnt[base + row];
  float scal = 1.0f / fmaxf(cv, 1.0f);
  const float* ap = agg + (size_t)(base + row) * HID;
  v16h ga0 = a_frag_global_scaled(ap, hi8, scal);
  v16h ga1 = a_frag_global_scaled(ap, 32 + hi8, scal);

  // ------------------- GRU: 12 output tiles of 16 cols (r,z,n) ------------
#pragma unroll
  for (int t = 0; t < 12; ++t) {
    const int col = t * 16 + row;
    v16h bi0 = b_frag(wtgih, col, hi8), bi1 = b_frag(wtgih, col, 32 + hi8);
    v16h bh0 = b_frag(wtghh, col, hi8), bh1 = b_frag(wtghh, col, 32 + hi8);
    v8f ai = {}, ah = {};
    ai = wmma_f16(ga0, bi0, ai); ai = wmma_f16(ga1, bi1, ai);
    ah = wmma_f16(xa0, bh0, ah); ah = wmma_f16(xa1, bh1, ah);
    const float bias_i = gbi[col];
    const float bias_h = gbh[col];
    if (t < 4) {                       // r gate -> bA
      const int hc = t * 16 + row;
#pragma unroll
      for (int v = 0; v < 8; ++v)
        bA[(mrow0 + v) * HID + hc] = sigm((ai[v] + bias_i) + (ah[v] + bias_h));
    } else if (t < 8) {                // z gate -> bB
      const int hc = (t - 4) * 16 + row;
#pragma unroll
      for (int v = 0; v < 8; ++v)
        bB[(mrow0 + v) * HID + hc] = sigm((ai[v] + bias_i) + (ah[v] + bias_h));
    } else {                           // n gate + combine -> h_conv into bA
      const int hc = (t - 8) * 16 + row;
#pragma unroll
      for (int v = 0; v < 8; ++v) {
        const int idx = (mrow0 + v) * HID + hc;
        float r  = bA[idx];
        float n  = htanh((ai[v] + bias_i) + r * (ah[v] + bias_h));
        float z  = bB[idx];
        float xv = xs[idx];
        bA[idx] = (1.0f - z) * n + z * xv;     // h_conv
      }
    }
  }

  // ------------------- LSTM (h0=c0=0): gates i, g, o (f unused) -----------
  v16h ha0 = a_frag_lds(bA, row, hi8);
  v16h ha1 = a_frag_lds(bA, row, 32 + hi8);
#pragma unroll
  for (int tt = 0; tt < 12; ++tt) {
    const int t = (tt < 4) ? tt : tt + 4;      // tiles 0-3 (i), 8-11 (g), 12-15 (o)
    const int col = t * 16 + row;
    v16h w0 = b_frag(wtlih, col, hi8), w1 = b_frag(wtlih, col, 32 + hi8);
    v8f acc = {};
    acc = wmma_f16(ha0, w0, acc); acc = wmma_f16(ha1, w1, acc);
    const float bs = bsum[col];
    const int hc = (t & 3) * 16 + row;
    if (t < 4) {                       // sig(i) -> bB
#pragma unroll
      for (int v = 0; v < 8; ++v)
        bB[(mrow0 + v) * HID + hc] = sigm(acc[v] + bs);
    } else if (t < 12) {               // c = sig(i)*tanh(g); tanh(c) -> xs
#pragma unroll
      for (int v = 0; v < 8; ++v) {
        const int idx = (mrow0 + v) * HID + hc;
        float c = bB[idx] * htanh(acc[v] + bs);
        xs[idx] = htanh(c);
      }
    } else {                           // h~ = sig(o)*tanh(c); relu -> bA
#pragma unroll
      for (int v = 0; v < 8; ++v) {
        const int idx = (mrow0 + v) * HID + hc;
        float ht = sigm(acc[v] + bs) * xs[idx];
        bA[idx] = fmaxf(ht, 0.0f);
      }
    }
  }

  // ------------------- final Linear ---------------------------------------
  v16h ra0 = a_frag_lds(bA, row, hi8);
  v16h ra1 = a_frag_lds(bA, row, 32 + hi8);
#pragma unroll
  for (int t = 0; t < 4; ++t) {
    const int col = t * 16 + row;
    v16h w0 = b_frag(wtlin, col, hi8), w1 = b_frag(wtlin, col, 32 + hi8);
    v8f acc = {};
    acc = wmma_f16(ra0, w0, acc); acc = wmma_f16(ra1, w1, acc);
    const float bb = linb[col];
#pragma unroll
    for (int v = 0; v < 8; ++v)
      out[(size_t)(base + mrow0 + v) * HID + col] = acc[v] + bb;
  }
}

// ---------------------------------------------------------------------------
extern "C" void kernel_launch(void* const* d_in, const int* in_sizes, int n_in,
                              void* d_out, int out_size, void* d_ws, size_t ws_size,
                              hipStream_t stream) {
  const float* x    = (const float*)d_in[0];
  const int*   ei   = (const int*)d_in[1];
  const float* Wc   = (const float*)d_in[2];
  const float* gih  = (const float*)d_in[3];
  const float* ghh  = (const float*)d_in[4];
  const float* gbi  = (const float*)d_in[5];
  const float* gbh  = (const float*)d_in[6];
  const float* lih  = (const float*)d_in[7];
  // d_in[8] = lstm_w_hh (unused: h0 = 0)
  const float* lbi  = (const float*)d_in[9];
  const float* lbh  = (const float*)d_in[10];
  const float* lin  = (const float*)d_in[11];
  const float* linb = (const float*)d_in[12];
  float* out = (float*)d_out;

  const int N = in_sizes[0] / HID;
  const int E = in_sizes[1] / 2;
  const int* src = ei;
  const int* dst = ei + E;

  // workspace layout (f32, then f16 region, 16B aligned)
  float* ws   = (float*)d_ws;
  float* m    = ws;
  float* agg  = m + (size_t)N * HID;
  float* cnt  = agg + (size_t)N * HID;
  float* bsum = cnt + N;
  _Float16* wtc   = (_Float16*)(bsum + 4 * HID);
  _Float16* wtgih = wtc + HID * HID;
  _Float16* wtghh = wtgih + 3 * HID * HID;
  _Float16* wtlih = wtghh + 3 * HID * HID;
  _Float16* wtlin = wtlih + 4 * HID * HID;

  const int ntiles = (N + 15) / 16;

  prep_kernel<<<64, 256, 0, stream>>>(Wc, gih, ghh, lih, lin, lbi, lbh,
                                      wtc, wtgih, wtghh, wtlih, wtlin, bsum);

  {
    long long total = (long long)N * HID + N;
    int blocks = (int)((total + 255) / 256);
    zero_kernel<<<blocks, 256, 0, stream>>>(agg, cnt, N);
  }

  conv_kernel<<<(ntiles + 7) / 8, 256, 0, stream>>>(x, wtc, m, ntiles);

  {
    long long threads = (long long)E * HID;
    int blocks = (int)((threads + 255) / 256);
    scatter_kernel<<<blocks, 256, 0, stream>>>(src, dst, m, agg, cnt, E);
  }

  node_kernel<<<(ntiles + 3) / 4, 128, 0, stream>>>(
      x, agg, cnt, wtgih, wtghh, wtlih, wtlin,
      gbi, gbh, bsum, linb, out, ntiles);
}
